// SimpleGCN_82806969467585
// MI455X (gfx1250) — compile-verified, matrix operand order checked by reading, not run on hardware
//
#include <hip/hip_runtime.h>

typedef __attribute__((ext_vector_type(16))) _Float16 v16h;
typedef __attribute__((ext_vector_type(8)))  float    v8f;

#define N_NODES 100000
#define N_EDGES 1600000
#define NGRAPH  64
#define FH      64     // F_IN == H == 64
#define GFEAT   32
#define GHID    32

// ---------------- degree / normalization ----------------
__global__ void k_init_deg(float* deg) {
    int i = blockIdx.x * blockDim.x + threadIdx.x;
    if (i < N_NODES) deg[i] = 1.0f;            // self loop
}

__global__ void k_edge_deg(const int* __restrict__ dst, float* deg) {
    int e = blockIdx.x * blockDim.x + threadIdx.x;
    if (e < N_EDGES) atomicAdd(&deg[dst[e]], 1.0f);
}

__global__ void k_deg_rsqrt(float* deg) {
    int i = blockIdx.x * blockDim.x + threadIdx.x;
    if (i < N_NODES) deg[i] = rsqrtf(deg[i]);
}

// ---------------- WMMA GEMM: out[N,64] = (relu?)h[N,64] @ W[64,64] ----------------
// B fragments are pre-swizzled into LDS in exact per-lane WMMA order so the
// inner loop is just ds_load_b128 pairs feeding v_wmma_f32_16x16x32_f16.
__global__ __launch_bounds__(256) void k_gemm_wmma(const float* __restrict__ h,
                                                   const float* __restrict__ W,
                                                   float* __restrict__ out,
                                                   int applyRelu) {
    // [tile t:4][chunk c:2][lane:32][elem:16] f16 = 8 KB
    __shared__ __align__(32) _Float16 Bl[4 * 2 * 32 * 16];
    int tid = threadIdx.x;
    for (int i = tid; i < 4096; i += 256) {
        int e    = i & 15;
        int lane = (i >> 4) & 31;
        int fc   = i >> 9;                     // t*2 + c
        int c    = fc & 1, t = fc >> 1;
        // fragment elem -> K: 0..7 -> khalf*8+e ; 8..15 -> 16+khalf*8+(e-8)
        int k    = ((e < 8) ? e : (e + 8)) + (lane >> 4) * 8 + c * 32;
        int n    = t * 16 + (lane & 15);
        Bl[i] = (_Float16)W[k * FH + n];
    }
    __syncthreads();

    int wave = tid >> 5;
    int lane = tid & 31;
    int r0 = (blockIdx.x * 8 + wave) * 16;     // 16-row tile per wave
    if (r0 >= N_NODES) return;                 // wave-uniform: EXEC stays all-1s

    int laneHi = lane >> 4;
    int c0     = laneHi * 8;                   // first K of this lane's segment
    int row    = r0 + (lane & 15);
    const float* hp = h + (long long)row * FH + c0;

    // A tile: contiguous 8-float runs -> vectorizable global_load_b128 pairs
    float buf0[16], buf1[16];
#pragma unroll
    for (int u = 0; u < 8; ++u) {
        buf0[u]     = hp[u];                   // K = c0 + u
        buf0[8 + u] = hp[16 + u];              // K = 16 + c0 + u
        buf1[u]     = hp[32 + u];              // K = 32 + c0 + u
        buf1[8 + u] = hp[48 + u];              // K = 48 + c0 + u
    }
    v16h a0, a1;
#pragma unroll
    for (int u = 0; u < 16; ++u) {
        float x = buf0[u], y = buf1[u];
        if (applyRelu) { x = fmaxf(x, 0.0f); y = fmaxf(y, 0.0f); }
        a0[u] = (_Float16)x;
        a1[u] = (_Float16)y;
    }

    const v16h* Bv = (const v16h*)Bl;          // 32 B per lane-fragment
    int ncol  = lane & 15;
    int mbase = r0 + laneHi * 8;               // D: VGPR v -> M=v (+8 upper lanes)
#pragma unroll
    for (int t = 0; t < 4; ++t) {              // four 16-column output tiles
        v16h b0 = Bv[(t * 2 + 0) * 32 + lane];
        v16h b1 = Bv[(t * 2 + 1) * 32 + lane];
        v8f c = {};
        c = __builtin_amdgcn_wmma_f32_16x16x32_f16(false, a0, false, b0,
                                                   (short)0, c, false, false);
        c = __builtin_amdgcn_wmma_f32_16x16x32_f16(false, a1, false, b1,
                                                   (short)0, c, false, false);
        int n0 = t * 16;
#pragma unroll
        for (int v = 0; v < 8; ++v)
            out[(long long)(mbase + v) * FH + n0 + ncol] = c[v];
    }
}

// ---------------- scatter: self-loop + bias init, then edge atomics ----------------
__global__ void k_scatter_init(const float* __restrict__ hw,
                               const float* __restrict__ dis,
                               const float* __restrict__ bias,
                               float* __restrict__ out) {
    int t = blockIdx.x * blockDim.x + threadIdx.x;
    int i = t >> 6, f = t & 63;
    if (i < N_NODES) {
        float d = dis[i];
        out[(long long)i * FH + f] = hw[(long long)i * FH + f] * d * d + bias[f];
    }
}

__global__ void k_scatter_edges(const float* __restrict__ hw,
                                const float* __restrict__ dis,
                                const int* __restrict__ src,
                                const int* __restrict__ dst,
                                float* __restrict__ out) {
    long long t = (long long)blockIdx.x * blockDim.x + threadIdx.x;
    int e = (int)(t >> 6), f = (int)(t & 63);
    if (e < N_EDGES) {
        int s = src[e], d = dst[e];
        float nrm = dis[s] * dis[d];
        atomicAdd(&out[(long long)d * FH + f], hw[(long long)s * FH + f] * nrm);
    }
}

// ---------------- pooling ----------------
__global__ void k_pool_zero(float* pooled) {
    int i = blockIdx.x * blockDim.x + threadIdx.x;
    if (i < NGRAPH * FH) pooled[i] = 0.0f;
}

__global__ void k_pool_sum(const float* __restrict__ h,
                           const int* __restrict__ gid,
                           float* __restrict__ pooled) {
    int t = blockIdx.x * blockDim.x + threadIdx.x;
    int i = t >> 6, f = t & 63;
    if (i < N_NODES) {
        float v = fmaxf(h[(long long)i * FH + f], 0.0f);   // relu of conv2 output
        atomicAdd(&pooled[gid[i] * FH + f], v);
    }
}

// ---------------- per-graph MLP head (tiny; 1 block, 1 thread per graph) ----------------
__global__ void k_head(const float* __restrict__ pooled,
                       const float* __restrict__ gfeat,
                       const float* __restrict__ glob_w, const float* __restrict__ glob_b,
                       const float* __restrict__ lin1_w, const float* __restrict__ lin1_b,
                       const float* __restrict__ lin2_w, const float* __restrict__ lin2_b,
                       const float* __restrict__ lin3_w, const float* __restrict__ lin3_b,
                       const float* __restrict__ lin4_w, const float* __restrict__ lin4_b,
                       float* __restrict__ out) {
    int g = threadIdx.x;
    if (g >= NGRAPH) return;

    float z[FH + GHID];
    for (int j = 0; j < FH; ++j) {                       // z1 = relu(pooled@lin1 + b)
        float acc = lin1_b[j];
        for (int k = 0; k < FH; ++k) acc += pooled[g * FH + k] * lin1_w[k * FH + j];
        z[j] = fmaxf(acc, 0.0f);
    }
    for (int j = 0; j < GHID; ++j) {                     // gx = relu(gf@glob + b)
        float acc = glob_b[j];
        for (int k = 0; k < GFEAT; ++k) acc += gfeat[g * GFEAT + k] * glob_w[k * GHID + j];
        z[FH + j] = fmaxf(acc, 0.0f);
    }
    float z2[FH];
    for (int j = 0; j < FH; ++j) {                       // z2 = relu([z1,gx]@lin2 + b)
        float acc = lin2_b[j];
        for (int k = 0; k < FH + GHID; ++k) acc += z[k] * lin2_w[k * FH + j];
        z2[j] = fmaxf(acc, 0.0f);
    }
    float z3[FH];
    for (int j = 0; j < FH; ++j) {                       // z3 = relu(z2@lin3 + b)
        float acc = lin3_b[j];
        for (int k = 0; k < FH; ++k) acc += z2[k] * lin3_w[k * FH + j];
        z3[j] = fmaxf(acc, 0.0f);
    }
    float acc = lin4_b[0];
    for (int k = 0; k < FH; ++k) acc += z3[k] * lin4_w[k];
    out[g] = acc;
}

extern "C" void kernel_launch(void* const* d_in, const int* in_sizes, int n_in,
                              void* d_out, int out_size, void* d_ws, size_t ws_size,
                              hipStream_t stream) {
    (void)in_sizes; (void)n_in; (void)out_size; (void)ws_size;

    const float* x       = (const float*)d_in[0];
    const float* gfeat   = (const float*)d_in[1];
    const float* conv_w[3] = {(const float*)d_in[2], (const float*)d_in[4], (const float*)d_in[6]};
    const float* conv_b[3] = {(const float*)d_in[3], (const float*)d_in[5], (const float*)d_in[7]};
    const float* lin1_w  = (const float*)d_in[8];
    const float* lin1_b  = (const float*)d_in[9];
    const float* glob_w  = (const float*)d_in[10];
    const float* glob_b  = (const float*)d_in[11];
    const float* lin2_w  = (const float*)d_in[12];
    const float* lin2_b  = (const float*)d_in[13];
    const float* lin3_w  = (const float*)d_in[14];
    const float* lin3_b  = (const float*)d_in[15];
    const float* lin4_w  = (const float*)d_in[16];
    const float* lin4_b  = (const float*)d_in[17];
    const int*   eidx    = (const int*)d_in[18];
    const int*   gid     = (const int*)d_in[19];
    const int*   esrc    = eidx;
    const int*   edst    = eidx + N_EDGES;
    float*       out     = (float*)d_out;

    // workspace carve-out (~51.6 MB)
    float* dis    = (float*)d_ws;                        // N
    float* bufA   = dis  + N_NODES;                      // N*64 : h / scatter accum
    float* bufB   = bufA + (long long)N_NODES * FH;      // N*64 : GEMM output
    float* pooled = bufB + (long long)N_NODES * FH;      // 64*64

    const int T = 256;
    int gN   = (N_NODES + T - 1) / T;
    int gE   = (N_EDGES + T - 1) / T;
    int gNF  = (N_NODES * FH) / T;             // 25000
    int gEF  = (N_EDGES * FH) / T;             // 400000
    int gGMM = (N_NODES / 16 + 7) / 8;         // 782 (8 waves/block, 16 rows/wave)

    // degree -> rsqrt
    k_init_deg <<<gN, T, 0, stream>>>(dis);
    k_edge_deg <<<gE, T, 0, stream>>>(edst, dis);
    k_deg_rsqrt<<<gN, T, 0, stream>>>(dis);

    // three GCN conv layers (ping-pong bufA <-> bufB)
    const float* hin = x;
    for (int l = 0; l < 3; ++l) {
        k_gemm_wmma    <<<gGMM, T, 0, stream>>>(hin, conv_w[l], bufB, l > 0 ? 1 : 0);
        k_scatter_init <<<gNF,  T, 0, stream>>>(bufB, dis, conv_b[l], bufA);
        k_scatter_edges<<<gEF,  T, 0, stream>>>(bufB, dis, esrc, edst, bufA);
        hin = bufA;
    }

    // per-graph sum pooling of relu(conv2 out)
    k_pool_zero<<<(NGRAPH * FH + T - 1) / T, T, 0, stream>>>(pooled);
    k_pool_sum <<<gNF, T, 0, stream>>>(bufA, gid, pooled);

    // MLP head
    k_head<<<1, 64, 0, stream>>>(pooled, gfeat, glob_w, glob_b,
                                 lin1_w, lin1_b, lin2_w, lin2_b,
                                 lin3_w, lin3_b, lin4_w, lin4_b, out);
}
